// Q_NsNet2_npy_90555090469150
// MI455X (gfx1250) — matrix operand order repaired and checked by reading, not compiled, not run on hardware
//
// ===================== CDNA5 / MI455X  Q-NsNet2 (round 3) =====================
// All GEMMs on v_wmma_f32_16x16x32_f16 with integer-coded (q-z) f16 operands
// (exact uint8 fake-quant math, scales folded into epilogues).
// B-operand weights stored transposed [N][K] -> both A and B fragments are
// 2x ds_load_b128 each (no repacking). Branch-free b128 global->LDS staging,
// double-buffered tiles. GRU recurrence: persistent 25-WGP kernel, Wh + h in
// LDS, global_load_async_to_lds_b128 h re-stage (ASYNCcnt), grid barrier/step.
#include <hip/hip_runtime.h>
#include <hip/hip_bf16.h>
#include <math.h>
#include <stdint.h>

typedef __attribute__((ext_vector_type(16))) _Float16 v16h;
typedef __attribute__((ext_vector_type(8)))  float    v8f;

#define DEV static __device__ __forceinline__

struct Q { float s, z; };
static inline Q qp(double mn, double mx) {
  Q q; q.s = (float)((mx - mn) / 255.0); q.z = roundf((float)(-mn) / q.s); return q;
}

DEV float fq_val(float x, float s, float z) {          // fake-quant -> real value
  float q = rintf(x / s) + z;
  q = fminf(fmaxf(q, 0.0f), 255.0f);
  return (q - z) * s;
}
DEV float fq_int(float x, float s, float z) {          // fake-quant -> (q - z)
  float q = rintf(x / s) + z;
  q = fminf(fmaxf(q, 0.0f), 255.0f);
  return q - z;
}
DEV float sigm(float x) { return 1.0f / (1.0f + __expf(-x)); }

// ---------------- WMMA fragment builders (16x16x32 f16) ----------------
// A 16x32 from row-major LDS tile: lane m=L&15, hs=L>>4 holds k hs*8..+7 and
// 16+hs*8..+7 -> two aligned b128 reads.
DEV v16h load_a_frag(const _Float16* __restrict__ p, int ld, int lane) {
  union { uint4 q[2]; v16h h; } r;
  int m = lane & 15, hs = lane >> 4;
  const _Float16* row = p + m * ld + hs * 8;
  r.q[0] = *(const uint4*)(row);
  r.q[1] = *(const uint4*)(row + 16);
  return r.h;
}
// B 32x16 from TRANSPOSED LDS tile ([n][k]): lane n=L&15, hs=L>>4 holds
// k hs*8..+7 and 16+hs*8..+7 -> two aligned b128 reads.
DEV v16h load_bt_frag(const _Float16* __restrict__ p, int ld, int lane) {
  union { uint4 q[2]; v16h h; } r;
  int n = lane & 15, hs = lane >> 4;
  const _Float16* row = p + n * ld + hs * 8;
  r.q[0] = *(const uint4*)(row);
  r.q[1] = *(const uint4*)(row + 16);
  return r.h;
}

// ---------------- elementwise quant prep (padded layouts) ----------------
__global__ void quant_pad_kernel(const float* __restrict__ src, _Float16* __restrict__ dst,
                                 long long srows, long long scols,
                                 long long drows, long long dld, float s, float z) {
  long long i = (long long)blockIdx.x * blockDim.x + threadIdx.x;
  if (i >= drows * dld) return;
  long long r = i / dld, c = i - r * dld;
  float v = 0.0f;
  if (r < srows && c < scols) v = fq_int(src[r * scols + c], s, z);
  dst[i] = (_Float16)v;
}
// weights: quantize + transpose ([K,N] row-major -> padded [N][K])
__global__ void quant_padT_kernel(const float* __restrict__ src, _Float16* __restrict__ dst,
                                  long long srows /*K*/, long long scols /*N*/,
                                  long long dn /*Npad*/, long long dk /*Kpad*/,
                                  float s, float z) {
  long long i = (long long)blockIdx.x * blockDim.x + threadIdx.x;
  if (i >= dn * dk) return;
  long long nrow = i / dk, k = i - nrow * dk;
  float v = 0.0f;
  if (k < srows && nrow < scols) v = fq_int(src[k * scols + nrow], s, z);
  dst[i] = (_Float16)v;
}
__global__ void fq_bias_kernel(const float* __restrict__ src, float* __restrict__ dst,
                               int n, float s, float z) {
  int i = blockIdx.x * blockDim.x + threadIdx.x;
  if (i < n) dst[i] = fq_val(src[i], s, z);
}

// ---------------- generic WMMA GEMM with fused fq epilogues ----------------
enum { EP_FC1 = 0, EP_ACE = 1, EP_RELU = 2, EP_SIG = 3 };

DEV void store_elem(float accv, long long orow, long long nn,
                    const float* __restrict__ bias, void* __restrict__ Cout,
                    long long ldc, long long coff, long long N, long long Npad,
                    float ab, int ep, float q1s, float q1z, float q2s, float q2z) {
  if (nn < N) {
    float x = accv * ab;
    if (ep == EP_FC1) {
      float v = fq_val(x, q1s, q1z) + bias[nn];
      ((_Float16*)Cout)[orow * ldc + coff + nn] = (_Float16)fq_int(v, q2s, q2z);
    } else if (ep == EP_ACE) {
      float v = fq_val(x, q1s, q1z) + bias[nn];
      ((_Float16*)Cout)[orow * ldc + coff + nn] = (_Float16)fq_val(v, q2s, q2z);
    } else if (ep == EP_RELU) {
      ((_Float16*)Cout)[orow * ldc + coff + nn] = (_Float16)fmaxf(x + bias[nn], 0.0f);
    } else {
      ((float*)Cout)[orow * ldc + coff + nn] = sigm(x + bias[nn]);
    }
  } else if (nn < Npad) {   // keep pad columns zero (buffer is a later GEMM A-operand)
    ((_Float16*)Cout)[orow * ldc + coff + nn] = (_Float16)0.0f;
  }
}

// M % 64 == 0; lda >= roundup32(K); Bw transposed [Npad][Kpad] with Kpad == roundup32(K),
// pads zeroed. All staging loads unguarded b128. trans: 0 none, 1 [B,T]->[T,B], 2 [T,B]->[B,T].
__launch_bounds__(256)
__global__ void gemm_wmma_f16(const _Float16* __restrict__ A, long long lda,
                              const _Float16* __restrict__ Bt, long long ldbt,
                              const float* __restrict__ bias,
                              void* __restrict__ Cout, long long ldc, long long coff,
                              long long M, long long N, long long Npad, long long K,
                              float ab, int ep,
                              float q1s, float q1z, float q2s, float q2z,
                              int trans) {
  __shared__ __align__(16) _Float16 As[2][64][40];   // 80B rows: 16B-aligned
  __shared__ __align__(16) _Float16 Bs[2][64][40];   // transposed tile [n][k]
  const int tid = threadIdx.x;
  const int lane = tid & 31;
  const int wave = tid >> 5;
  const int wm = wave & 3;            // row tile 0..3
  const int ct0 = (wave >> 2) * 2;    // col tiles {0,1} or {2,3}
  const long long m0 = (long long)blockIdx.y * 64;
  const long long n0 = (long long)blockIdx.x * 64;

  // staging coordinates (one 16B chunk per thread per tile, both operands)
  const int sr = tid >> 2, sc = (tid & 3) * 8;   // 64 rows x 4 chunks
  const _Float16* Aptr = A + (m0 + sr) * lda + sc;
  const _Float16* Bptr = Bt + (n0 + sr) * ldbt + sc;

  const int nk = (int)((K + 31) / 32);
  {
    uint4 a0 = *(const uint4*)Aptr;
    uint4 b0 = *(const uint4*)Bptr;
    *(uint4*)&As[0][sr][sc] = a0;
    *(uint4*)&Bs[0][sr][sc] = b0;
  }
  v8f acc0 = {}; v8f acc1 = {};
  for (int kt = 0; kt < nk; ++kt) {
    __syncthreads();
    const int cur = kt & 1;
    if (kt + 1 < nk) {                 // stage next tile while computing current
      uint4 an = *(const uint4*)(Aptr + (long long)(kt + 1) * 32);
      uint4 bn = *(const uint4*)(Bptr + (long long)(kt + 1) * 32);
      *(uint4*)&As[cur ^ 1][sr][sc] = an;
      *(uint4*)&Bs[cur ^ 1][sr][sc] = bn;
    }
    v16h a  = load_a_frag(&As[cur][wm * 16][0], 40, lane);
    v16h b0 = load_bt_frag(&Bs[cur][ct0 * 16][0], 40, lane);
    v16h b1 = load_bt_frag(&Bs[cur][ct0 * 16 + 16][0], 40, lane);
    acc0 = __builtin_amdgcn_wmma_f32_16x16x32_f16(false, a, false, b0, (short)0, acc0, false, false);
    acc1 = __builtin_amdgcn_wmma_f32_16x16x32_f16(false, a, false, b1, (short)0, acc1, false, false);
  }
#pragma unroll
  for (int i = 0; i < 8; ++i) {
    int mi = (int)m0 + wm * 16 + (lane >> 4) * 8 + i;
    long long orow;
    if (trans == 1)      { int bb = mi / 1000, tt = mi - bb * 1000; orow = (long long)tt * 64 + bb; }
    else if (trans == 2) { int tt = mi >> 6,   bb = mi & 63;        orow = (long long)bb * 1000 + tt; }
    else                 orow = mi;
    long long nA = n0 + ct0 * 16 + (lane & 15);
    store_elem(acc0[i], orow, nA,      bias, Cout, ldc, coff, N, Npad, ab, ep, q1s, q1z, q2s, q2z);
    store_elem(acc1[i], orow, nA + 16, bias, Cout, ldc, coff, N, Npad, ab, ep, q1s, q1z, q2s, q2z);
  }
}

// ---------------- persistent GRU recurrence ----------------
#define GH 400
#define GB 64
#define GKP 416      // K padded to 13*32
#define NWG 25
#define GCOLS 16     // H columns per workgroup
#define WTLD 416     // row stride (k) of transposed Wh [448][416]
#define HSLD 448     // padded row stride of hs outputs

struct GruQ {
  float s_h;                       // scale of incoming h (rnn{L}GRU)
  float s_wh0, s_wh1, s_wh2;       // Whr/Whz/Whn scales
  Q pre0, pre1, pre2;              // b_/d_/f_
  Q post0, post1, post2;           // b/d/f
  Q r_, r, z_, z, n1, n2, nq, hn1, hn2, hn3, hnew;
};

DEV void async_lds_load16(void* lds_dst, const void* gsrc) {
  unsigned l = (unsigned)(uintptr_t)lds_dst;     // flat LDS addr low 32b == LDS offset
  asm volatile("global_load_async_to_lds_b128 %0, %1, off"
               :: "v"(l), "v"(gsrc) : "memory");
}

DEV void grid_barrier(unsigned* cnt, unsigned* gen, unsigned nwg) {
  __threadfence();
  __syncthreads();
  if (threadIdx.x == 0) {
    unsigned g = __hip_atomic_load(gen, __ATOMIC_RELAXED, __HIP_MEMORY_SCOPE_AGENT);
    unsigned a = __hip_atomic_fetch_add(cnt, 1u, __ATOMIC_ACQ_REL, __HIP_MEMORY_SCOPE_AGENT);
    if (a == nwg - 1u) {
      __hip_atomic_store(cnt, 0u, __ATOMIC_RELAXED, __HIP_MEMORY_SCOPE_AGENT);
      __hip_atomic_store(gen, g + 1u, __ATOMIC_RELEASE, __HIP_MEMORY_SCOPE_AGENT);
    } else {
      while (__hip_atomic_load(gen, __ATOMIC_ACQUIRE, __HIP_MEMORY_SCOPE_AGENT) == g)
        __builtin_amdgcn_s_sleep(8);
    }
  }
  __syncthreads();
  __threadfence();
}

__launch_bounds__(384)
__global__ void gru_recurrent(const _Float16* __restrict__ inproj,   // [T,64,1200] real f16 (r|z|n)
                              const _Float16* __restrict__ Whr,
                              const _Float16* __restrict__ Whz,
                              const _Float16* __restrict__ Whn,      // transposed [448][416] int f16
                              const float* __restrict__ bhr,
                              const float* __restrict__ bhz,
                              const float* __restrict__ bhn,        // fq'd real biases
                              _Float16* __restrict__ h_buf,         // 2 x [64*400] int f16 ping-pong
                              _Float16* __restrict__ hs_out,        // [T,64,HSLD] int f16
                              unsigned* __restrict__ ctrl, int T, GruQ q) {
  extern __shared__ __align__(16) char smem[];
  _Float16* h_lds  = (_Float16*)smem;                                  // [64][416]
  _Float16* wh_lds = (_Float16*)(smem + GB * GKP * 2);                 // [3][16 n][416 k]
  float*    gate_lds = (float*)(smem + GB * GKP * 2 + 3 * GCOLS * GKP * 2); // [3][64][16]

  const int tid = threadIdx.x;
  const int lane = tid & 31;
  const int wave = tid >> 5;       // 12 waves
  const int g = wave >> 2;         // gate 0..2 (r,z,n)
  const int mt = wave & 3;         // 16-row tile of B=64
  const int n0 = blockIdx.x * GCOLS;

  const float* bh = (g == 0) ? bhr : ((g == 1) ? bhz : bhn);
  const float swh = (g == 0) ? q.s_wh0 : ((g == 1) ? q.s_wh1 : q.s_wh2);
  const Q pre     = (g == 0) ? q.pre0  : ((g == 1) ? q.pre1  : q.pre2);
  const Q post    = (g == 0) ? q.post0 : ((g == 1) ? q.post1 : q.post2);
  const float bhv = bh[n0 + (lane & 15)];

  // resident Wh slice: 3 gates x 16 n-rows x 416 k, b128 chunks (transposed source)
  for (int c = tid; c < 3 * GCOLS * (GKP / 8); c += 384) {
    int gg = c / (GCOLS * (GKP / 8)), rem = c - gg * GCOLS * (GKP / 8);
    int j = rem / (GKP / 8), kc = (rem - j * (GKP / 8)) * 8;
    const _Float16* W = (gg == 0) ? Whr : ((gg == 1) ? Whz : Whn);
    uint4 v = *(const uint4*)(W + (long long)(n0 + j) * WTLD + kc);
    *(uint4*)&wh_lds[(gg * GCOLS + j) * GKP + kc] = v;
  }
  // zero h_lds K-pad columns
  for (int e = tid; e < GB * (GKP - GH) / 8; e += 384) {
    int m = e / 2, c = GH + (e & 1) * 8;
    uint4 zz = {0, 0, 0, 0};
    *(uint4*)&h_lds[m * GKP + c] = zz;
  }
  __syncthreads();

  for (int t = 0; t < T; ++t) {
    // ---- stage h_t into LDS via CDNA5 async global->LDS (16B chunks) ----
    const _Float16* hsrc = h_buf + (size_t)(t & 1) * GB * GH;
    for (int chunk = tid; chunk < (GB * GH) / 8; chunk += 384) {
      int idx = chunk * 8;
      int m = idx / GH, c = idx - m * GH;             // 8 | 400, chunks never cross rows
      async_lds_load16(&h_lds[m * GKP + c], hsrc + idx);
    }
    asm volatile("s_wait_asynccnt 0" ::: "memory");
    __syncthreads();

    // ---- gate GEMM: (h @ Wh_g)[mt*16.., n0..n0+16], fully unrolled K ----
    v8f acc = {};
    const _Float16* hrow = h_lds + mt * 16 * GKP;
    const _Float16* wgp = wh_lds + g * GCOLS * GKP;
#pragma unroll
    for (int kk = 0; kk < GKP / 32; ++kk) {
      v16h a = load_a_frag(hrow + kk * 32, GKP, lane);
      v16h b = load_bt_frag(wgp + kk * 32, GKP, lane);
      acc = __builtin_amdgcn_wmma_f32_16x16x32_f16(false, a, false, b, (short)0, acc, false, false);
    }
    const float sv = q.s_h * swh;
#pragma unroll
    for (int i = 0; i < 8; ++i) {
      int ml = mt * 16 + (lane >> 4) * 8 + i;
      int nl = lane & 15;
      float v = fq_val(acc[i] * sv, pre.s, pre.z) + bhv;
      gate_lds[(g * GB + ml) * GCOLS + nl] = fq_val(v, post.s, post.z);
    }
    __syncthreads();

    // ---- GRU elementwise (gate-0 waves cover all 64 rows) ----
    if (g == 0) {
#pragma unroll
      for (int i = 0; i < 8; ++i) {
        int ml = mt * 16 + (lane >> 4) * 8 + i;
        int nl = lane & 15;
        size_t ib = ((size_t)t * GB + ml) * 1200 + n0 + nl;
        float av = (float)inproj[ib];
        float cv = (float)inproj[ib + 400];
        float ev = (float)inproj[ib + 800];
        float bv = gate_lds[(0 * GB + ml) * GCOLS + nl];
        float dv = gate_lds[(1 * GB + ml) * GCOLS + nl];
        float fv = gate_lds[(2 * GB + ml) * GCOLS + nl];
        float rg = fq_val(sigm(fq_val(av + bv, q.r_.s, q.r_.z)), q.r.s, q.r.z);
        float zg = fq_val(sigm(fq_val(cv + dv, q.z_.s, q.z_.z)), q.z.s, q.z.z);
        float ng = fq_val(tanhf(fq_val(ev + fq_val(rg * fv, q.n1.s, q.n1.z), q.n2.s, q.n2.z)),
                          q.nq.s, q.nq.z);
        float hr  = (float)h_lds[ml * GKP + n0 + nl] * q.s_h;
        float h1v = fq_val(1.0f - zg, q.hn1.s, q.hn1.z);
        float h2v = fq_val(h1v * ng, q.hn2.s, q.hn2.z);
        float h3v = fq_val(zg * hr, q.hn3.s, q.hn3.z);
        _Float16 hq = (_Float16)fq_int(h2v + h3v, q.hnew.s, q.hnew.z);
        h_buf[(size_t)((t + 1) & 1) * GB * GH + ml * GH + n0 + nl] = hq;
        hs_out[((size_t)t * GB + ml) * HSLD + n0 + nl] = hq;
      }
    }
    grid_barrier(ctrl, ctrl + 1, NWG);   // h_{t+1} globally visible before next stage
  }
}

// ===================== host side =====================
extern "C" void kernel_launch(void* const* d_in, const int* in_sizes, int n_in,
                              void* d_out, int out_size, void* d_ws, size_t ws_size,
                              hipStream_t stream) {
  (void)in_sizes; (void)n_in; (void)out_size; (void)ws_size;
  const int B = 64, T = 1000, F = 257, H = 400, D = 600;
  const long long MBT = (long long)B * T;
  // padded dims
  const long long FP = 288;                 // roundup32(257)
  const long long HKP = 416, HNP = 448;     // H as K / as N
  const long long DKP = 608, DNP = 640;     // D as K / as N
  const long long FNP = 320;                // roundup64(257)

  const float* x    = (const float*)d_in[0];
  const float* Wfc1 = (const float*)d_in[1];
  const float* bfc1 = (const float*)d_in[2];
  const float* Wiz1 = (const float*)d_in[3],  *Whz1 = (const float*)d_in[4];
  const float* biz1 = (const float*)d_in[5],  *bhz1 = (const float*)d_in[6];
  const float* Wir1 = (const float*)d_in[7],  *Whr1 = (const float*)d_in[8];
  const float* bir1 = (const float*)d_in[9],  *bhr1 = (const float*)d_in[10];
  const float* Win1 = (const float*)d_in[11], *Whn1 = (const float*)d_in[12];
  const float* bin1 = (const float*)d_in[13], *bhn1 = (const float*)d_in[14];
  const float* Wiz2 = (const float*)d_in[15], *Whz2 = (const float*)d_in[16];
  const float* biz2 = (const float*)d_in[17], *bhz2 = (const float*)d_in[18];
  const float* Wir2 = (const float*)d_in[19], *Whr2 = (const float*)d_in[20];
  const float* bir2 = (const float*)d_in[21], *bhr2 = (const float*)d_in[22];
  const float* Win2 = (const float*)d_in[23], *Whn2 = (const float*)d_in[24];
  const float* bin2 = (const float*)d_in[25], *bhn2 = (const float*)d_in[26];
  const float* Wfc2 = (const float*)d_in[27], *bfc2 = (const float*)d_in[28];
  const float* Wfc3 = (const float*)d_in[29], *bfc3 = (const float*)d_in[30];
  const float* Wfc4 = (const float*)d_in[31], *bfc4 = (const float*)d_in[32];

  // -------- calibration params --------
  Q qx_   = qp(-0.0025095, 0.0022181);
  Q q166  = qp(-0.22075387835502625, 0.208940327167511);
  Q qm1   = qp(-0.00291599917, 0.0017367251);
  Q qb1   = qp(-0.48688140511512756, 0.5176185369491577);
  Q qa1   = qp(-0.48778465390205383, 0.5181604027748108);
  Q qWiz1 = qp(-0.43284985423088074, 0.46175122261047363);
  Q qWir1 = qp(-0.34401071071624756, 0.29191476106643677);
  Q qWin1 = qp(-0.3236880302429199, 0.39607325196266174);
  Q qWhz1 = qp(-1.8417714834213257, 1.7173254489898682);
  Q qWhr1 = qp(-1.1574513912200928, 1.0300449132919312);
  Q qWhn1 = qp(-0.7756922245025635, 0.9530389308929443);
  Q qbiz1 = qp(-0.5063393712043762, 0.36664387583732605);
  Q qbir1 = qp(-0.07920225709676743, 0.20611026883125305);
  Q qbin1 = qp(-0.5539973378181458, 0.17938342690467834);
  Q qbhz1 = qp(-0.5337516665458679, 0.4148772358894348);
  Q qbhr1 = qp(-0.07688436657190323, 0.14814253151416779);
  Q qbhn1 = qp(-0.7828555107116699, 0.9008108973503113);
  Q qWiz2 = qp(-0.9102030992507935, 0.9408696889877319);
  Q qWir2 = qp(-0.9560997486114502, 0.6683358550071716);
  Q qWin2 = qp(-0.4721935987472534, 0.48561596870422363);
  Q qWhz2 = qp(-1.2992678880691528, 1.2991048097610474);
  Q qWhr2 = qp(-0.8318714499473572, 1.1085889339447021);
  Q qWhn2 = qp(-0.955470085144043, 1.046797513961792);
  Q qbiz2 = qp(-0.44805487990379333, 0.1560053527355194);
  Q qbir2 = qp(-0.08767592161893845, 0.11347303539514542);
  Q qbin2 = qp(-0.239909827709198, 0.12033259868621826);
  Q qbhz2 = qp(-0.43745461106300354, 0.12699371576309204);
  Q qbhr2 = qp(-0.09617264568805695, 0.07690174877643585);
  Q qbhn2 = qp(-0.17204178869724274, 0.19739042222499847);
  Q q207  = qp(-1.3657219409942627, 1.158295750617981);
  Q qfb2  = qp(-0.1750922054052353, 0.1385071724653244);
  Q q208  = qp(-3.1666038036346436, 2.5026357173919678);
  Q qfb3  = qp(-0.10188056528568268, 0.0899151861667633);
  Q q209  = qp(-1.300571084022522, 1.928941249847412);
  Q qfb4  = qp(-0.10699586570262909, 0.04597663879394531);
  Q g1a_ = qp(-0.6389939785003662, 0.7715625762939453),  g1a = qp(-0.6046768426895142, 0.8871182203292847);
  Q g1c_ = qp(-1.5660111904144287, 1.0454494953155518),  g1c = qp(-1.9779117107391357, 1.3700535297393799);
  Q g1e_ = qp(-1.3637111186981201, 1.018247127532959),   g1e = qp(-1.8583884239196777, 1.1587692499160767);
  Q g1b_ = qp(-0.004922409541904926, 0.004103424027562141), g1b = qp(-0.07475128024816513, 0.14630259573459625);
  Q g1d_ = qp(-0.008429044857621193, 0.006403823848813772), g1d = qp(-0.5529640316963196, 0.41507571935653687);
  Q g1f_ = qp(-0.005411399528384209, 0.0061536869034171104), g1f = qp(-0.7833794355392456, 0.8978855013847351);
  Q g1r_ = qp(-0.6226556301116943, 0.9800534844398499),  g1r = qp(0.3491777181625366, 0.7271188497543335);
  Q g1z_ = qp(-2.392332077026367, 1.7851293087005615),   g1z = qp(0.08375928550958633, 0.856329083442688);
  Q g1n1 = qp(-0.5516456365585327, 0.4556601643562317);
  Q g1n2 = qp(-2.4100341796875, 1.1423156261444092);
  Q g1n  = qp(-0.983996570110321, 0.8151924014091492);
  Q g1h1 = qp(0.143670916557312, 0.9162406921386719);
  Q g1h2 = qp(-0.9015777111053467, 0.45391541719436646);
  Q g1h3 = qp(-0.0016954239690676332, 0.0016671211924403906);
  Q grnn = qp(-0.9016271829605103, 0.4546271562576294);  // rnn1GRU == rnn2GRU
  Q g2a_ = qp(-0.681461751461029, 0.9113116264343262),   g2a = qp(-0.7149235010147095, 0.908741295337677);
  Q g2c_ = qp(-1.2279887199401855, 1.1102137565612793),  g2c = qp(-1.6760436296463013, 1.200895071029663);
  Q g2e_ = qp(-0.6732944250106812, 0.6314664483070374),  g2e = g1e;
  Q g2b_ = qp(-0.005148397758603096, 0.011014967225492), g2b = qp(-0.09436552226543427, 0.07623167335987091);
  Q g2d_ = qp(-0.008141756057739258, 0.00894666276872158), g2d = qp(-0.43875735998153687, 0.1262134611606598);

  // -------- workspace layout (padded strides; weights transposed [N][K]) --------
  char* base = (char*)d_ws;
  size_t off = 0;
  auto alloc = [&](size_t bytes) -> void* {
    void* p = base + off; off += (bytes + 255) & ~(size_t)255; return p;
  };
  unsigned* ctrl  = (unsigned*)alloc(256);
  _Float16* h_pp  = (_Float16*)alloc((size_t)2 * B * H * 2);
  _Float16* qxb   = (_Float16*)alloc((size_t)MBT * FP * 2);      // [64000,288]
  _Float16* wF1   = (_Float16*)alloc((size_t)HNP * FP * 2);      // [448,288]
  _Float16* wIr1  = (_Float16*)alloc((size_t)HNP * HKP * 2);     // [448,416]
  _Float16* wIz1  = (_Float16*)alloc((size_t)HNP * HKP * 2);
  _Float16* wIn1  = (_Float16*)alloc((size_t)HNP * HKP * 2);
  _Float16* wHr1  = (_Float16*)alloc((size_t)HNP * HKP * 2);
  _Float16* wHz1  = (_Float16*)alloc((size_t)HNP * HKP * 2);
  _Float16* wHn1  = (_Float16*)alloc((size_t)HNP * HKP * 2);
  _Float16* wIr2  = (_Float16*)alloc((size_t)HNP * HKP * 2);
  _Float16* wIz2  = (_Float16*)alloc((size_t)HNP * HKP * 2);
  _Float16* wIn2  = (_Float16*)alloc((size_t)HNP * HKP * 2);
  _Float16* wHr2  = (_Float16*)alloc((size_t)HNP * HKP * 2);
  _Float16* wHz2  = (_Float16*)alloc((size_t)HNP * HKP * 2);
  _Float16* wHn2  = (_Float16*)alloc((size_t)HNP * HKP * 2);
  _Float16* wF2   = (_Float16*)alloc((size_t)DNP * HKP * 2);     // [640,416]
  _Float16* wF3   = (_Float16*)alloc((size_t)DNP * DKP * 2);     // [640,608]
  _Float16* wF4   = (_Float16*)alloc((size_t)FNP * DKP * 2);     // [320,608]
  float* rbF1 = (float*)alloc(H * 4);
  float* rbIr1 = (float*)alloc(H * 4); float* rbIz1 = (float*)alloc(H * 4); float* rbIn1 = (float*)alloc(H * 4);
  float* rbHr1 = (float*)alloc(H * 4); float* rbHz1 = (float*)alloc(H * 4); float* rbHn1 = (float*)alloc(H * 4);
  float* rbIr2 = (float*)alloc(H * 4); float* rbIz2 = (float*)alloc(H * 4); float* rbIn2 = (float*)alloc(H * 4);
  float* rbHr2 = (float*)alloc(H * 4); float* rbHz2 = (float*)alloc(H * 4); float* rbHn2 = (float*)alloc(H * 4);
  float* rbF2 = (float*)alloc(D * 4); float* rbF3 = (float*)alloc(D * 4); float* rbF4 = (float*)alloc(F * 4);
  _Float16* xsb  = (_Float16*)alloc((size_t)MBT * HNP * 2);      // [T,B,448] int (fc1Add scale)
  _Float16* ipr  = (_Float16*)alloc((size_t)MBT * 1200 * 2);     // [T,B,1200] real (r|z|n)
  _Float16* hs1  = (_Float16*)alloc((size_t)MBT * HNP * 2);      // [T,B,448]
  _Float16* hs2  = (_Float16*)alloc((size_t)MBT * HNP * 2);
  _Float16* y2   = (_Float16*)alloc((size_t)MBT * DNP * 2);      // [T*B,640]
  _Float16* y3   = (_Float16*)alloc((size_t)MBT * DNP * 2);

  hipMemsetAsync(ctrl, 0, 256, stream);
  hipMemsetAsync(h_pp, 0, (size_t)2 * B * H * 2, stream);
  hipMemsetAsync(hs1, 0, (size_t)MBT * HNP * 2, stream);   // zero pad cols (A-operand later)
  hipMemsetAsync(hs2, 0, (size_t)MBT * HNP * 2, stream);

  auto quantA = [&](const float* s, _Float16* d, long long sr, long long sc,
                    long long dr, long long dld, Q q) {
    long long n = dr * dld;
    quant_pad_kernel<<<dim3((unsigned)((n + 255) / 256)), dim3(256), 0, stream>>>(s, d, sr, sc, dr, dld, q.s, q.z);
  };
  auto quantWT = [&](const float* s, _Float16* d, long long K_, long long N_,
                     long long Np, long long Kp, Q q) {
    long long n = Np * Kp;
    quant_padT_kernel<<<dim3((unsigned)((n + 255) / 256)), dim3(256), 0, stream>>>(s, d, K_, N_, Np, Kp, q.s, q.z);
  };
  auto quantB = [&](const float* s, float* d, int n, Q q) {
    fq_bias_kernel<<<dim3((unsigned)((n + 255) / 256)), dim3(256), 0, stream>>>(s, d, n, q.s, q.z);
  };
  quantA(x, qxb, MBT, F, MBT, FP, qx_);
  quantWT(Wfc1, wF1, F, H, HNP, FP, q166);
  quantWT(Wir1, wIr1, H, H, HNP, HKP, qWir1); quantWT(Wiz1, wIz1, H, H, HNP, HKP, qWiz1);
  quantWT(Win1, wIn1, H, H, HNP, HKP, qWin1);
  quantWT(Whr1, wHr1, H, H, HNP, HKP, qWhr1); quantWT(Whz1, wHz1, H, H, HNP, HKP, qWhz1);
  quantWT(Whn1, wHn1, H, H, HNP, HKP, qWhn1);
  quantWT(Wir2, wIr2, H, H, HNP, HKP, qWir2); quantWT(Wiz2, wIz2, H, H, HNP, HKP, qWiz2);
  quantWT(Win2, wIn2, H, H, HNP, HKP, qWin2);
  quantWT(Whr2, wHr2, H, H, HNP, HKP, qWhr2); quantWT(Whz2, wHz2, H, H, HNP, HKP, qWhz2);
  quantWT(Whn2, wHn2, H, H, HNP, HKP, qWhn2);
  quantWT(Wfc2, wF2, H, D, DNP, HKP, q207);
  quantWT(Wfc3, wF3, D, D, DNP, DKP, q208);
  quantWT(Wfc4, wF4, D, F, FNP, DKP, q209);
  quantB(bfc1, rbF1, H, qb1);
  quantB(bir1, rbIr1, H, qbir1); quantB(biz1, rbIz1, H, qbiz1); quantB(bin1, rbIn1, H, qbin1);
  quantB(bhr1, rbHr1, H, qbhr1); quantB(bhz1, rbHz1, H, qbhz1); quantB(bhn1, rbHn1, H, qbhn1);
  quantB(bir2, rbIr2, H, qbir2); quantB(biz2, rbIz2, H, qbiz2); quantB(bin2, rbIn2, H, qbin2);
  quantB(bhr2, rbHr2, H, qbhr2); quantB(bhz2, rbHz2, H, qbhz2); quantB(bhn2, rbHn2, H, qbhn2);
  quantB(bfc2, rbF2, D, qfb2); quantB(bfc3, rbF3, D, qfb3); quantB(bfc4, rbF4, F, qfb4);

  auto gemm = [&](const _Float16* A, long long lda, const _Float16* Bt, long long ldbt,
                  const float* bias, void* C, long long ldc, long long coff,
                  long long M, long long N, long long Npad, long long K, float ab, int ep,
                  Q q1, Q q2, int trans) {
    dim3 grid((unsigned)((N + 63) / 64), (unsigned)(M / 64));
    gemm_wmma_f16<<<grid, dim3(256), 0, stream>>>(A, lda, Bt, ldbt, bias, C, ldc, coff,
                                                  M, N, Npad, K, ab, ep, q1.s, q1.z, q2.s, q2.z,
                                                  trans);
  };
  Q qz0 = {1.0f, 0.0f};

  // fc1: [B*T,257] x [257,400] -> xs [T,B,448] (int, fc1Add scale; pads zeroed)
  gemm(qxb, FP, wF1, FP, rbF1, xsb, HNP, 0, MBT, H, HNP, F, qx_.s * q166.s, EP_FC1, qm1, qa1, 1);

  // layer-1 input projections -> ipr [T,B, r|z|n]
  gemm(xsb, HNP, wIr1, HKP, rbIr1, ipr, 1200, 0,   MBT, H, H, H, qa1.s * qWir1.s, EP_ACE, g1a_, g1a, 0);
  gemm(xsb, HNP, wIz1, HKP, rbIz1, ipr, 1200, 400, MBT, H, H, H, qa1.s * qWiz1.s, EP_ACE, g1c_, g1c, 0);
  gemm(xsb, HNP, wIn1, HKP, rbIn1, ipr, 1200, 800, MBT, H, H, H, qa1.s * qWin1.s, EP_ACE, g1e_, g1e, 0);

  GruQ G1;
  G1.s_h = grnn.s; G1.s_wh0 = qWhr1.s; G1.s_wh1 = qWhz1.s; G1.s_wh2 = qWhn1.s;
  G1.pre0 = g1b_; G1.pre1 = g1d_; G1.pre2 = g1f_;
  G1.post0 = g1b; G1.post1 = g1d; G1.post2 = g1f;
  G1.r_ = g1r_; G1.r = g1r; G1.z_ = g1z_; G1.z = g1z;
  G1.n1 = g1n1; G1.n2 = g1n2; G1.nq = g1n;
  G1.hn1 = g1h1; G1.hn2 = g1h2; G1.hn3 = g1h3; G1.hnew = grnn;
  size_t shmem = (size_t)GB * GKP * 2 + 3 * GCOLS * GKP * 2 + 3 * GB * GCOLS * 4;
  gru_recurrent<<<dim3(NWG), dim3(384), shmem, stream>>>(ipr, wHr1, wHz1, wHn1,
      rbHr1, rbHz1, rbHn1, h_pp, hs1, ctrl, T, G1);

  // layer-2 input projections (A = hs1, scale rnn1GRU)
  gemm(hs1, HNP, wIr2, HKP, rbIr2, ipr, 1200, 0,   MBT, H, H, H, grnn.s * qWir2.s, EP_ACE, g2a_, g2a, 0);
  gemm(hs1, HNP, wIz2, HKP, rbIz2, ipr, 1200, 400, MBT, H, H, H, grnn.s * qWiz2.s, EP_ACE, g2c_, g2c, 0);
  gemm(hs1, HNP, wIn2, HKP, rbIn2, ipr, 1200, 800, MBT, H, H, H, grnn.s * qWin2.s, EP_ACE, g2e_, g2e, 0);

  hipMemsetAsync(h_pp, 0, (size_t)2 * B * H * 2, stream);   // h0 = 0 for layer 2
  GruQ G2 = G1;
  G2.s_wh0 = qWhr2.s; G2.s_wh1 = qWhz2.s; G2.s_wh2 = qWhn2.s;
  G2.pre0 = g2b_; G2.pre1 = g2d_; G2.pre2 = g1f_;
  G2.post0 = g2b; G2.post1 = g2d; G2.post2 = g1f;
  gru_recurrent<<<dim3(NWG), dim3(384), shmem, stream>>>(ipr, wHr2, wHz2, wHn2,
      rbHr2, rbHz2, rbHn2, h_pp, hs2, ctrl, T, G2);

  // fc2 / fc3 (relu, real f16 through the chain), fc4 (sigmoid -> d_out [B,T,257])
  gemm(hs2, HNP, wF2, HKP, rbF2, y2, DNP, 0, MBT, D, DNP, H, grnn.s * q207.s, EP_RELU, qz0, qz0, 0);
  gemm(y2,  DNP, wF3, DKP, rbF3, y3, DNP, 0, MBT, D, DNP, D, q208.s,          EP_RELU, qz0, qz0, 0);
  gemm(y3,  DNP, wF4, DKP, rbF4, d_out, F, 0, MBT, F, F, D, q209.s,           EP_SIG,  qz0, qz0, 2);
}
